// BlockEqLinear_50087908606417
// MI455X (gfx1250) — matrix operand order repaired and checked by reading, not compiled, not run on hardware
//
#include <hip/hip_runtime.h>

// ---------------------------------------------------------------------------
// BlockEqLinear on MI455X (gfx1250): bf16 WMMA + async-to-LDS double buffering.
//   out[b,g,k,p] = sum_h x[b,k,h]*(A-B)[g,p,h] + sum_h S[b,h]*B[g,p,h]
// Stage 1: W1=bf16(A-B), W2=bf16(B)                         (ws)
// Stage 2: Xbf = bf16(x)                                    (ws)
// Stage 3: Sbf[b][h] = bf16(sum_k x[b,k,h])                 (ws)
// Stage 4: P = Sbf @ W2^T                 (f32 ws)          [gemm<false>]
// Stage 5: out = Xbf @ W1^T + P (scatter)                   [gemm<true>]
// ---------------------------------------------------------------------------

typedef __bf16         v16bf __attribute__((ext_vector_type(16)));
typedef float          v8f   __attribute__((ext_vector_type(8)));
typedef unsigned int   u32x4 __attribute__((ext_vector_type(4)));
typedef int            i32x4 __attribute__((ext_vector_type(4)));
typedef float          f32x4 __attribute__((ext_vector_type(4)));

typedef __attribute__((address_space(1))) i32x4 g_i32x4;
typedef __attribute__((address_space(3))) i32x4 l_i32x4;

union FragU { u32x4 u[2]; v16bf v; };

// CDNA5 async global->LDS path (ASYNCcnt-tracked), with sync fallback.
#if defined(__has_builtin)
#if __has_builtin(__builtin_amdgcn_global_load_async_to_lds_b128) && \
    __has_builtin(__builtin_amdgcn_s_wait_asynccnt)
#define USE_ASYNC 1
#endif
#endif
#ifndef USE_ASYNC
#define USE_ASYNC 0
#endif

__device__ __forceinline__ void cp_g2l_b128(const unsigned short* g, unsigned short* l) {
#if USE_ASYNC
    __builtin_amdgcn_global_load_async_to_lds_b128(
        (g_i32x4*)g, (l_i32x4*)l, 0, 0);
#else
    *(u32x4*)l = *(const u32x4*)g;
#endif
}
#if USE_ASYNC
#define ASYNC_WAIT(n) __builtin_amdgcn_s_wait_asynccnt(n)
#else
#define ASYNC_WAIT(n)
#endif

__device__ __forceinline__ unsigned short f32_to_bf16(float f) {
    unsigned u = __float_as_uint(f);
    unsigned r = 0x7FFFu + ((u >> 16) & 1u);   // round-to-nearest-even
    return (unsigned short)((u + r) >> 16);
}
__device__ __forceinline__ unsigned pack_bf16x2(float f0, float f1) {
    return (unsigned)f32_to_bf16(f0) | ((unsigned)f32_to_bf16(f1) << 16);
}

// ---------------- Stage 1: weight prep -------------------------------------
__global__ __launch_bounds__(256) void prep_w_kernel(
    const float* __restrict__ A, const float* __restrict__ B,
    unsigned short* __restrict__ W1, unsigned short* __restrict__ W2)
{
    int idx = blockIdx.x * 256 + threadIdx.x;     // 4*512*512 = 1048576
    float a = A[idx], b = B[idx];
    W1[idx] = f32_to_bf16(a - b);
    W2[idx] = f32_to_bf16(b);
}

// ---------------- Stage 2: x -> bf16 (vectorized, 8 elems/thread) ----------
__global__ __launch_bounds__(256) void conv_x_kernel(
    const float* __restrict__ x, unsigned short* __restrict__ Xbf)
{
    int idx = blockIdx.x * 256 + threadIdx.x;     // 16777216/8 = 2097152
    const f32x4* s = (const f32x4*)(x + (size_t)idx * 8);
    f32x4 a = s[0], b = s[1];
    u32x4 o = { pack_bf16x2(a[0], a[1]), pack_bf16x2(a[2], a[3]),
                pack_bf16x2(b[0], b[1]), pack_bf16x2(b[2], b[3]) };
    *(u32x4*)(Xbf + (size_t)idx * 8) = o;
}

// ---------------- Stage 3: k-sum -> bf16 -----------------------------------
__global__ __launch_bounds__(256) void sum_k_kernel(
    const float* __restrict__ x, unsigned short* __restrict__ Sbf)
{
    int idx = blockIdx.x * 256 + threadIdx.x;     // 4096*512 = 2097152
    int b = idx >> 9, h = idx & 511;
    const float* row = x + (size_t)b * 4096 + h;
    float s = 0.f;
#pragma unroll
    for (int k = 0; k < 8; ++k) s += row[k * 512];
    Sbf[idx] = f32_to_bf16(s);
}

// ---------------- Stages 4/5: tiled bf16 WMMA GEMM, async double-buffer ----
#define BM 128
#define BN 128
#define BK 32
#define NTOT 2048
#define LDT 40     // LDS row stride (halfwords): 80B, 16B-aligned, conflict-free

template <bool FUSED>
__global__ __launch_bounds__(256) void gemm_bf16_kernel(
    const unsigned short* __restrict__ Xsrc,  // [M][512] bf16 (Xbf or Sbf)
    const unsigned short* __restrict__ W,     // [2048][512] bf16
    const float*          __restrict__ Cmat,  // FUSED: P[M/8][2048], else null
    float*                __restrict__ Out)
{
    __shared__ unsigned short sA[2][BM * LDT];
    __shared__ unsigned short sB[2][BN * LDT];

    const int tid   = threadIdx.x;
    const int lane  = tid & 31;
    const int wave  = tid >> 5;
    const int wm    = wave & 3;     // 4 waves along M, 32 rows each
    const int wn    = wave >> 2;    // 2 waves along N, 64 cols each
    const int mBase = blockIdx.y * BM;
    const int nBase = blockIdx.x * BN;

    const int mofs  = (lane & 16) ? 8 : 0;  // C/D layout: lanes 16-31 -> M+8
    const int nlane = lane & 15;            // C/D layout: N = lane&15

    v8f acc[2][4];
#pragma unroll
    for (int i = 0; i < 2; ++i)
#pragma unroll
        for (int j = 0; j < 4; ++j) {
            if (FUSED) {
                const int n = nBase + wn * 64 + j * 16 + nlane;
#pragma unroll
                for (int r = 0; r < 8; ++r) {
                    int m = mBase + wm * 32 + i * 16 + mofs + r;
                    acc[i][j][r] = Cmat[(size_t)(m >> 3) * NTOT + n];
                }
            } else {
#pragma unroll
                for (int r = 0; r < 8; ++r) acc[i][j][r] = 0.f;
            }
        }

    const int srow = tid >> 1;          // 0..127
    const int sseg = (tid & 1) * 16;    // 0 or 16 (halfwords)

    // Per-thread staging of one 32-byte row segment of each tile:
    // 4 async b128 ops per tile per thread.
    const unsigned short* gX = Xsrc + (size_t)(mBase + srow) * 512 + sseg;
    const unsigned short* gW = W    + (size_t)(nBase + srow) * 512 + sseg;
    const int lofs = srow * LDT + sseg;

    auto stage = [&](int kk, int buf) {
        cp_g2l_b128(gX + kk,     &sA[buf][lofs]);
        cp_g2l_b128(gX + kk + 8, &sA[buf][lofs + 8]);
        cp_g2l_b128(gW + kk,     &sB[buf][lofs]);
        cp_g2l_b128(gW + kk + 8, &sB[buf][lofs + 8]);
    };

    stage(0, 0);
    for (int t = 0; t < 16; ++t) {
        const int cur = t & 1;
        if (t < 15) {
            stage((t + 1) * BK, cur ^ 1);   // prefetch next tile (other buffer)
            ASYNC_WAIT(4);                  // tile t complete; t+1 in flight
        } else {
            ASYNC_WAIT(0);
        }
        __syncthreads();                    // tile t visible to all waves

        // A fragments: 16x32 bf16 (ISA 7.12.2): lane m=lane&15,
        // K in {ko..ko+7, 16+ko..16+ko+7}, ko=(lane&16)?8:0
        FragU a0, a1;
        {
            const int ko = (lane & 16) ? 8 : 0;
            const unsigned short* p0 = &sA[cur][(wm * 32 + nlane) * LDT + ko];
            a0.u[0] = *(const u32x4*)p0;
            a0.u[1] = *(const u32x4*)(p0 + 16);
            const unsigned short* p1 = &sA[cur][(wm * 32 + 16 + nlane) * LDT + ko];
            a1.u[0] = *(const u32x4*)p1;
            a1.u[1] = *(const u32x4*)(p1 + 16);
        }
        // B fragments: 32x16 bf16: lane n=lane&15, K in {ko..ko+7, 16+ko..16+ko+7}, ko=(lane&16)?8:0
        FragU bf[4];
#pragma unroll
        for (int j = 0; j < 4; ++j) {
            const int ko = (lane & 16) ? 8 : 0;
            const unsigned short* p = &sB[cur][(wn * 64 + j * 16 + nlane) * LDT + ko];
            bf[j].u[0] = *(const u32x4*)p;
            bf[j].u[1] = *(const u32x4*)(p + 16);
        }

#pragma unroll
        for (int j = 0; j < 4; ++j) {
            acc[0][j] = __builtin_amdgcn_wmma_f32_16x16x32_bf16(
                false, a0.v, false, bf[j].v, (short)0, acc[0][j], false, false);
            acc[1][j] = __builtin_amdgcn_wmma_f32_16x16x32_bf16(
                false, a1.v, false, bf[j].v, (short)0, acc[1][j], false, false);
        }
        __syncthreads();    // all waves done reading before buffer reuse
    }

    // Store D. FUSED: scatter to out[b,g,k,p] = [(m>>3), (n>>9), (m&7), (n&511)]
#pragma unroll
    for (int i = 0; i < 2; ++i)
#pragma unroll
        for (int j = 0; j < 4; ++j) {
            const int n = nBase + wn * 64 + j * 16 + nlane;
#pragma unroll
            for (int r = 0; r < 8; ++r) {
                int m = mBase + wm * 32 + i * 16 + mofs + r;
                float v = acc[i][j][r];
                if (FUSED) {
                    Out[(size_t)(m >> 3) * 16384 + (size_t)(n >> 9) * 4096
                        + (size_t)(m & 7) * 512 + (n & 511)] = v;
                } else {
                    Out[(size_t)m * NTOT + n] = v;
                }
            }
        }
}

// ---------------------------------------------------------------------------
extern "C" void kernel_launch(void* const* d_in, const int* in_sizes, int n_in,
                              void* d_out, int out_size, void* d_ws, size_t ws_size,
                              hipStream_t stream) {
    (void)in_sizes; (void)n_in; (void)out_size; (void)ws_size;
    const float* x = (const float*)d_in[0];   // [4096][4096]
    const float* A = (const float*)d_in[1];   // [4][512][512]
    const float* B = (const float*)d_in[2];   // [4][512][512]
    float* out = (float*)d_out;               // [4096][16384]

    // Workspace layout (72 MiB total):
    char* ws = (char*)d_ws;
    unsigned short* W1  = (unsigned short*)ws;                     //  2 MiB @ 0
    unsigned short* W2  = (unsigned short*)(ws + (2u  << 20));     //  2 MiB
    unsigned short* Sbf = (unsigned short*)(ws + (4u  << 20));     //  4 MiB
    unsigned short* Xbf = (unsigned short*)(ws + (8u  << 20));     // 32 MiB
    float*          P   = (float*)         (ws + (40u << 20));     // 32 MiB

    prep_w_kernel<<<4096, 256, 0, stream>>>(A, B, W1, W2);
    conv_x_kernel<<<8192, 256, 0, stream>>>(x, Xbf);
    sum_k_kernel <<<8192, 256, 0, stream>>>(x, Sbf);

    // P = S @ W2^T : M=4096, N=2048, K=512
    gemm_bf16_kernel<false><<<dim3(16, 32),  256, 0, stream>>>(Sbf, W2, nullptr, P);
    // out = X @ W1^T + P : M=32768, N=2048, K=512
    gemm_bf16_kernel<true> <<<dim3(16, 256), 256, 0, stream>>>(Xbf, W1, P, out);
}
